// Contrastive_loss_30915174596795
// MI455X (gfx1250) — compile-verified, matrix operand order checked by reading, not run on hardware
//
#include <hip/hip_runtime.h>
#include <hip/hip_bf16.h>
#include <stdint.h>

// ---------------------------------------------------------------------------
// Contrastive (InfoNCE-style) loss on MI455X / gfx1250.
// Compute-bound: ~103 GFLOP of 4096x4096x256 GEMMs vs ~34MB of input traffic,
// so everything funnels through v_wmma_f32_16x16x32_bf16 with fp32 accum.
// Similarity matrices are never materialized: exp + row/col reduction happen
// in the WMMA accumulator registers.
// Staging uses CDNA5 GLOBAL_LOAD_ASYNC_TO_LDS_B128 (ASYNCcnt) with a
// double-buffered LDS tile so global fetch overlaps the WMMA pipeline.
// Each wave owns a 32x128 strip (2 A-frags x 8 N-tiles = 16 accumulators) so
// every B fragment read from LDS feeds two WMMAs (1.25 ds_loads per WMMA).
// ---------------------------------------------------------------------------

typedef __attribute__((ext_vector_type(8)))  __bf16 v8bf;
typedef __attribute__((ext_vector_type(16))) __bf16 v16bf;
typedef __attribute__((ext_vector_type(8)))  float  v8f;

union Frag16 { v16bf v; v8bf h[2]; };

constexpr int BATCH = 4;
constexpr int HW    = 4096;   // rows per batch (64*64)
constexpr int CH    = 256;    // feature dim (K)
constexpr float INV_TAU = 2.5f;                 // 1 / 0.4
constexpr float E_SELF  = 12.182493960703473f;  // exp(1/tau), diag of refl

// -- CDNA5 async global->LDS copy (16 bytes per lane), tracked by ASYNCcnt --
__device__ __forceinline__ void async_ld16(uint32_t lds_addr, const void* gaddr) {
  asm volatile("global_load_async_to_lds_b128 %0, %1, off"
               :: "v"(lds_addr), "v"((uint64_t)(uintptr_t)gaddr)
               : "memory");
}
__device__ __forceinline__ void wait_async_le(int n) {
  if (n == 0)
    asm volatile("s_wait_asynccnt 0x0" ::: "memory");
  else
    asm volatile("s_wait_asynccnt 0x6" ::: "memory");
}

// ---------------------------------------------------------------------------
// Kernel 1: L2-normalize each projected row and store as bf16 [b, n, c].
// Input z is [b, c, hw]; element (b,n,c) lives at z[(b*CH + c)*HW + n].
// ---------------------------------------------------------------------------
__global__ void normalize_kernel(const float* __restrict__ z,
                                 __bf16* __restrict__ out) {
  int r = blockIdx.x * blockDim.x + threadIdx.x;   // 0 .. BATCH*HW-1
  int b = r >> 12;
  int n = r & (HW - 1);
  const float* zb = z + (size_t)b * CH * HW + n;
  float ss = 0.f;
  for (int c = 0; c < CH; ++c) {
    float v = zb[(size_t)c * HW];
    ss += v * v;
  }
  float rn = 1.0f / fmaxf(sqrtf(ss), 1e-12f);
  __bf16* o = out + (size_t)r * CH;
  for (int c = 0; c < CH; ++c) {
    o[c] = (__bf16)(zb[(size_t)c * HW] * rn);
  }
}

// ---------------------------------------------------------------------------
// Kernel 2: S = X * Y^T per batch, fused exp(S/tau) + row sums (+ col sums).
// 256x128 tile per 256-thread workgroup: 8 waves, each wave owns a 32x128
// strip = 2x8 16x16 WMMA accumulators. K=256 swept in 32-wide stages through
// a double-buffered LDS tile filled by async global->LDS DMA.
//
// Fragment layout (CDNA5 ISA 7.12.2, 16-bit A 16x32):
//   lanes 0-15  : M = lane,    K = {0..7, 16..23}
//   lanes 16-31 : M = lane-16, K = {8..15, 24..31}
// B (K x N = Y^T) mirrors this with N = lane%16, so both fragments are two
// 16-byte DS loads from the staged row-major tiles.
// C/D layout: VGPR r holds (M = r + 8*(lane>=16), N = lane%16).
// ---------------------------------------------------------------------------
__global__ __launch_bounds__(256)
void gemm_exp_sum_kernel(const __bf16* __restrict__ X,
                         const __bf16* __restrict__ Y,
                         float* __restrict__ rowsum,
                         float* __restrict__ colsum) {
  __shared__ __align__(16) __bf16 xs[2][256 * 32];   // 2 x 16KB
  __shared__ __align__(16) __bf16 ys[2][128 * 32];   // 2 x  8KB

  const int b  = blockIdx.z;
  const int m0 = blockIdx.y * 256;
  const int n0 = blockIdx.x * 128;
  const __bf16* Xb = X + (size_t)b * HW * CH;
  const __bf16* Yb = Y + (size_t)b * HW * CH;

  const int t    = threadIdx.x;
  const int wave = t >> 5;
  const int lane = t & 31;
  const int lrow = lane & 15;
  const int koff = (lane < 16) ? 0 : 8;

  // staging assignment: per stage each thread DMA-copies one full X row
  // segment (32 bf16 = 4 x b128) and a half Y row (16 bf16 = 2 x b128)
  const __bf16* xsrc = &Xb[(size_t)(m0 + t) * CH];
  const __bf16* ysrc = &Yb[(size_t)(n0 + (t >> 1)) * CH + (t & 1) * 16];

  v8f acc[2][8] = {};

  auto issue_stage = [&](int buf, int k0) {
    uint32_t xl = (uint32_t)(uintptr_t)&xs[buf][t * 32];
    uint32_t yl = (uint32_t)(uintptr_t)&ys[buf][(t >> 1) * 32 + (t & 1) * 16];
    async_ld16(xl,      xsrc + k0);
    async_ld16(xl + 16, xsrc + k0 + 8);
    async_ld16(xl + 32, xsrc + k0 + 16);
    async_ld16(xl + 48, xsrc + k0 + 24);
    async_ld16(yl,      ysrc + k0);
    async_ld16(yl + 16, ysrc + k0 + 8);
  };

  constexpr int NSTAGE = CH / 32;   // 8
  issue_stage(0, 0);

  for (int i = 0; i < NSTAGE; ++i) {
    const int cur = i & 1;
    if (i + 1 < NSTAGE) {
      issue_stage((i + 1) & 1, (i + 1) * 32);   // prefetch next stage
      wait_async_le(6);                         // stage i landed (in-order)
    } else {
      wait_async_le(0);
    }
    __syncthreads();

    Frag16 a0, a1;
    a0.h[0] = *(const v8bf*)&xs[cur][(wave * 32 + lrow) * 32 + koff];
    a0.h[1] = *(const v8bf*)&xs[cur][(wave * 32 + lrow) * 32 + koff + 16];
    a1.h[0] = *(const v8bf*)&xs[cur][(wave * 32 + 16 + lrow) * 32 + koff];
    a1.h[1] = *(const v8bf*)&xs[cur][(wave * 32 + 16 + lrow) * 32 + koff + 16];

#pragma unroll
    for (int nt = 0; nt < 8; ++nt) {
      Frag16 bfrag;
      bfrag.h[0] = *(const v8bf*)&ys[cur][(nt * 16 + lrow) * 32 + koff];
      bfrag.h[1] = *(const v8bf*)&ys[cur][(nt * 16 + lrow) * 32 + koff + 16];
      acc[0][nt] = __builtin_amdgcn_wmma_f32_16x16x32_bf16(
          false, a0.v, false, bfrag.v, (short)0, acc[0][nt], false, false);
      acc[1][nt] = __builtin_amdgcn_wmma_f32_16x16x32_bf16(
          false, a1.v, false, bfrag.v, (short)0, acc[1][nt], false, false);
    }
    __syncthreads();   // stage i fully consumed; buffer free for stage i+2
  }

  // Epilogue: e = exp(s/tau); reduce rows across N-lanes, cols across M.
  float rowpart[2][8];
#pragma unroll
  for (int mt = 0; mt < 2; ++mt)
#pragma unroll
    for (int r = 0; r < 8; ++r) rowpart[mt][r] = 0.f;

#pragma unroll
  for (int nt = 0; nt < 8; ++nt) {
    float cpart = 0.f;
#pragma unroll
    for (int mt = 0; mt < 2; ++mt) {
#pragma unroll
      for (int r = 0; r < 8; ++r) {
        float e = __expf(acc[mt][nt][r] * INV_TAU);
        rowpart[mt][r] += e;
        cpart += e;
      }
    }
    if (colsum) {
      // lanes l and l^16 hold the two M-halves of the same column
      cpart += __shfl_xor(cpart, 16, 32);
      if (lane < 16)
        atomicAdd(&colsum[(size_t)b * HW + n0 + nt * 16 + lane], cpart);
    }
  }

#pragma unroll
  for (int mt = 0; mt < 2; ++mt) {
#pragma unroll
    for (int r = 0; r < 8; ++r) {
      float v = rowpart[mt][r];
      v += __shfl_xor(v, 1, 32);
      v += __shfl_xor(v, 2, 32);
      v += __shfl_xor(v, 4, 32);
      v += __shfl_xor(v, 8, 32);
      if (lrow == 0) {
        int m = m0 + wave * 32 + mt * 16 + r + ((lane >> 4) << 3);
        atomicAdd(&rowsum[(size_t)b * HW + m], v);
      }
    }
  }
}

// ---------------------------------------------------------------------------
// Kernel 3: diagonal dot products + final loss reduction.
//   l1[n] = -log(diag / (rs_aa + rs_ab - e^{1/tau}))
//   l2[n] = -log(diag / (rs_bb + cs_ab - e^{1/tau}))   (cs_ab == rowsum of S_ba)
//   out   = mean(0.5*(l1+l2))
// One wave per row; 8 rows per 256-thread block.
// ---------------------------------------------------------------------------
__global__ void loss_kernel(const __bf16* __restrict__ A,
                            const __bf16* __restrict__ Bm,
                            const float* __restrict__ rs_aa,
                            const float* __restrict__ rs_ab,
                            const float* __restrict__ cs_ab,
                            const float* __restrict__ rs_bb,
                            float* __restrict__ out) {
  const int wave = threadIdx.x >> 5;
  const int lane = threadIdx.x & 31;
  const int r = blockIdx.x * 8 + wave;   // 0 .. BATCH*HW-1

  const v8bf av = *(const v8bf*)&A[(size_t)r * CH + lane * 8];
  const v8bf bv = *(const v8bf*)&Bm[(size_t)r * CH + lane * 8];
  float dot = 0.f;
#pragma unroll
  for (int i = 0; i < 8; ++i) dot += (float)av[i] * (float)bv[i];
  dot += __shfl_xor(dot, 1, 32);
  dot += __shfl_xor(dot, 2, 32);
  dot += __shfl_xor(dot, 4, 32);
  dot += __shfl_xor(dot, 8, 32);
  dot += __shfl_xor(dot, 16, 32);

  if (lane == 0) {
    float diag = __expf(dot * INV_TAU);
    float l1 = -__logf(diag / (rs_aa[r] + rs_ab[r] - E_SELF));
    float l2 = -__logf(diag / (rs_bb[r] + cs_ab[r] - E_SELF));
    atomicAdd(out, (l1 + l2) * 0.5f / (float)(BATCH * HW));
  }
}

// ---------------------------------------------------------------------------
// Launch: normalize -> 3 fused GEMM/exp/sum passes -> loss.
// Workspace: Abf(8MB) | Bbf(8MB) | rs_aa | rs_ab | cs_ab | rs_bb (64KB each).
// ---------------------------------------------------------------------------
extern "C" void kernel_launch(void* const* d_in, const int* in_sizes, int n_in,
                              void* d_out, int out_size, void* d_ws, size_t ws_size,
                              hipStream_t stream) {
  const float* z1 = (const float*)d_in[0];
  const float* z2 = (const float*)d_in[1];
  float* out = (float*)d_out;

  char* ws = (char*)d_ws;
  const size_t bf_bytes = (size_t)BATCH * HW * CH * sizeof(__bf16);   // 8 MB
  __bf16* Abf = (__bf16*)ws;
  __bf16* Bbf = (__bf16*)(ws + bf_bytes);
  float* rs_aa = (float*)(ws + 2 * bf_bytes);
  float* rs_ab = rs_aa + (size_t)BATCH * HW;
  float* cs_ab = rs_ab + (size_t)BATCH * HW;
  float* rs_bb = cs_ab + (size_t)BATCH * HW;

  // zero the atomic accumulators + output (graph-capture-safe stream memsets)
  hipMemsetAsync(rs_aa, 0, 4 * (size_t)BATCH * HW * sizeof(float), stream);
  hipMemsetAsync(out, 0, sizeof(float), stream);

  normalize_kernel<<<BATCH * HW / 256, 256, 0, stream>>>(z1, Abf);
  normalize_kernel<<<BATCH * HW / 256, 256, 0, stream>>>(z2, Bbf);

  dim3 grid(HW / 128, HW / 256, BATCH);
  // refl(A,A): row sums only (S_aa symmetric; only rows needed)
  gemm_exp_sum_kernel<<<grid, 256, 0, stream>>>(Abf, Abf, rs_aa, nullptr);
  // between(A,B): row sums feed l1, column sums are S_ba row sums -> l2
  gemm_exp_sum_kernel<<<grid, 256, 0, stream>>>(Abf, Bbf, rs_ab, cs_ab);
  // refl(B,B)
  gemm_exp_sum_kernel<<<grid, 256, 0, stream>>>(Bbf, Bbf, rs_bb, nullptr);

  loss_kernel<<<BATCH * HW / 8, 256, 0, stream>>>(Abf, Bbf, rs_aa, rs_ab,
                                                  cs_ab, rs_bb, out);
}
